// GraphTransformer_240518168864
// MI455X (gfx1250) — matrix operand order repaired and checked by reading, not compiled, not run on hardware
//
#include <hip/hip_runtime.h>
#include <hip/hip_bf16.h>

// ---------------------------------------------------------------------------
// GraphTransformer (3x TransformerConv), MI455X / gfx1250, wave32 + WMMA f16
// GEMM: async global->LDS double-buffered B panels + v_wmma_f32_16x16x32_f16
// ---------------------------------------------------------------------------
#define N_NODES   10000
#define E_EDGES   320000
#define DIM       256          // IN_DIM == HD == heads*d
#define HEADS     8
#define HDIM      32           // per-head dim
#define ROWTILES  (N_NODES/16) // 625
#define KTILES    (DIM/32)     // 8
#define STAGES    8            // 4 matrices x 2 column halves
#define STAGE_F16 32768        // 64 panels * 512 f16 = 64KB per stage
#define LDS_BYTES (2 * STAGE_F16 * 2)   // 128KB double buffer

typedef __attribute__((ext_vector_type(16))) _Float16 v16h;
typedef __attribute__((ext_vector_type(8)))  _Float16 v8h;
typedef __attribute__((ext_vector_type(8)))  float    v8f;

struct Ptrs8 { const float* p[8]; };

// ----- monotone float<->uint key for atomic segment-max --------------------
__device__ __forceinline__ unsigned fkey(float f) {
    unsigned b = __float_as_uint(f);
    return (b & 0x80000000u) ? ~b : (b | 0x80000000u);
}
__device__ __forceinline__ float funkey(unsigned k) {
    unsigned b = (k & 0x80000000u) ? (k & 0x7FFFFFFFu) : ~k;
    return __uint_as_float(b);
}

// ----- pack 8 weight matrices into stage-major B panels --------------------
// Layer layout: stage s = (mat*2 + half), 64 panels of 512 f16 each,
// panel (kt, ntl): element [lane*16+s] =
//   W[(kt*32 + (lane<16?0:16) + s)*256 + (half*8+ntl)*16 + (lane&15)]
__global__ void pack_w(Ptrs8 wp, _Float16* __restrict__ out) {
    int t = blockIdx.x * blockDim.x + threadIdx.x;     // 8*65536 total
    int mi  = t >> 16;
    int r   = t & 65535;
    int P   = r >> 9;           // panel 0..127 within matrix
    int e   = r & 511;
    int lane = e >> 4;
    int s    = e & 15;
    int ntl  = P & 7;
    int kt   = (P >> 3) & 7;
    int half = (P >> 6) & 1;
    int n = (half * 8 + ntl) * 16 + (lane & 15);
    int k = kt * 32 + ((lane < 16) ? 0 : 16) + s;
    out[t] = (_Float16)wp.p[mi][k * DIM + n];
}

// ----- pack 8 bias vectors (256 each) into one contiguous buffer -----------
__global__ void pack_bias(Ptrs8 bp, float* __restrict__ out) {
    int t = blockIdx.x * blockDim.x + threadIdx.x;     // 8*256 total
    out[t] = bp.p[t >> 8][t & 255];
}

// ----- activation f32 -> f16 ----------------------------------------------
__global__ void to_f16(const float* __restrict__ in, _Float16* __restrict__ out) {
    int t = blockIdx.x * blockDim.x + threadIdx.x;
    out[t] = (_Float16)in[t];
}

// ----- fused 4-way GEMM: [q|k|v|xr](stacked) = h @ {Wq,Wk,Wv,Wskip} + bias -
// One block (8 waves) per 16-row stripe. B panels stream through a 2x64KB
// LDS double buffer via async global->LDS B128 copies (ASYNCcnt), WMMAs read
// B from LDS (ds_load) -> long-latency global traffic fully overlapped.
__global__ __launch_bounds__(256, 1)
void gemm4(const _Float16* __restrict__ A16,
           const _Float16* __restrict__ wpk,   // 4 packed matrices, stage-major
           const float* __restrict__ bias,     // 4*256
           float* __restrict__ out) {          // 4*N*256
    extern __shared__ _Float16 lds[];          // 2 * STAGE_F16
    const int tid  = threadIdx.x;
    const int lane = tid & 31;
    const int wave = tid >> 5;                 // 0..7
    const int rowtile = blockIdx.x;            // 0..624

    // ---- A fragments: 16 rows x 256 cols, kept in VGPRs (L2-resident) ----
    const int row = rowtile * 16 + (lane & 15);
    const int kb  = (lane < 16) ? 0 : 16;
    const _Float16* arow = A16 + (size_t)row * DIM;
    v16h afrag[KTILES];
#pragma unroll
    for (int kt = 0; kt < KTILES; ++kt) {
        v8h lo = *(const v8h*)(arow + kt * 32 + kb);
        v8h hi = *(const v8h*)(arow + kt * 32 + kb + 8);
        afrag[kt] = __builtin_shufflevector(lo, hi, 0,1,2,3,4,5,6,7,8,9,10,11,12,13,14,15);
    }

    // ---- async bulk copy of one 64KB weight stage into LDS buffer `buf` ---
    auto issue_stage = [&](int s, int buf) {
        const char* g = (const char*)wpk + (size_t)s * (STAGE_F16 * 2);
        unsigned lbase = (unsigned)buf * (STAGE_F16 * 2);   // dynamic LDS at offset 0
#pragma unroll
        for (int i = 0; i < 16; ++i) {
            unsigned off = (unsigned)(i * 256 + tid) * 16;
            unsigned laddr = lbase + off;
            const char* gaddr = g + off;
            asm volatile("global_load_async_to_lds_b128 %0, %1, off"
                         :: "v"(laddr), "v"(gaddr) : "memory");
        }
    };

    issue_stage(0, 0);
    asm volatile("s_wait_asynccnt 0x0" ::: "memory");
    __syncthreads();

#pragma unroll
    for (int s = 0; s < STAGES; ++s) {
        const int buf  = s & 1;
        const int mat  = s >> 1;
        const int half = s & 1;
        if (s + 1 < STAGES) issue_stage(s + 1, buf ^ 1);   // prefetch next stage

        // ---- compute: each wave does one 16x16 output tile from LDS B ----
        const _Float16* sb = lds + buf * STAGE_F16;
        v8f c = {};
#pragma unroll
        for (int kt = 0; kt < KTILES; ++kt) {
            v16h b = *(const v16h*)(sb + (size_t)(kt * 8 + wave) * 512 + lane * 16);
            c = __builtin_amdgcn_wmma_f32_16x16x32_f16(false, afrag[kt], false, b,
                                                       (short)0, c, false, false);
        }

        // ---- epilogue: bias + store ----
        int nt  = half * 8 + wave;
        int col = nt * 16 + (lane & 15);
        float bv = bias[mat * DIM + col];
        float* op = out + (size_t)mat * N_NODES * DIM + col;
        int rbase = rowtile * 16 + ((lane < 16) ? 0 : 8);
#pragma unroll
        for (int r = 0; r < 8; ++r)
            op[(size_t)(rbase + r) * DIM] = c[r] + bv;

        if (s + 1 < STAGES) {
            asm volatile("s_wait_asynccnt 0x0" ::: "memory");
            __syncthreads();
        }
    }
}

// ----- per-layer init: out=0, mkey=0 (== -inf key floor), denom=0 ----------
__global__ void init_layer(float* __restrict__ aggout, unsigned* __restrict__ mkey,
                           float* __restrict__ denom) {
    int t = blockIdx.x * blockDim.x + threadIdx.x;
    if (t < N_NODES * DIM) aggout[t] = 0.f;
    if (t < N_NODES * HEADS) { mkey[t] = 0u; denom[t] = 0.f; }
}

// ----- pass 1: alpha + segment max -----------------------------------------
__global__ void edge_alpha(const int* __restrict__ ei, const float* __restrict__ ew,
                           const float* __restrict__ q, const float* __restrict__ k,
                           const float* __restrict__ We,
                           float* __restrict__ alpha, unsigned* __restrict__ mkey) {
    int t = blockIdx.x * blockDim.x + threadIdx.x;  // E*8
    int e = t >> 3, h = t & 7;
    int src = ei[e], dst = ei[E_EDGES + e];
    float w = ew[e];
    const float* qp = q + (size_t)dst * DIM + h * HDIM;
    const float* kp = k + (size_t)src * DIM + h * HDIM;
    const float* wp = We + h * HDIM;
    float acc = 0.f;
#pragma unroll
    for (int d = 0; d < HDIM; ++d)
        acc += qp[d] * (kp[d] + w * wp[d]);
    acc *= 0.17677669529663688f;                    // 1/sqrt(32)
    alpha[t] = acc;
    atomicMax(mkey + (size_t)dst * HEADS + h, fkey(acc));
}

// ----- pass 2: exp(alpha - m) + segment sum --------------------------------
__global__ void edge_exp(const int* __restrict__ ei, float* __restrict__ alpha,
                         const unsigned* __restrict__ mkey, float* __restrict__ denom) {
    int t = blockIdx.x * blockDim.x + threadIdx.x;  // E*8
    int e = t >> 3, h = t & 7;
    int dst = ei[E_EDGES + e];
    float m = funkey(mkey[(size_t)dst * HEADS + h]);
    if ((__float_as_uint(m) & 0x7F800000u) == 0x7F800000u) m = 0.f;  // !isfinite -> 0
    float ex = __expf(alpha[t] - m);
    alpha[t] = ex;
    atomicAdd(denom + (size_t)dst * HEADS + h, ex);
}

// ----- pass 3: weighted scatter-add of (v[src] + e) ------------------------
__global__ void edge_agg(const int* __restrict__ ei, const float* __restrict__ ew,
                         const float* __restrict__ v, const float* __restrict__ We,
                         const float* __restrict__ alpha, const float* __restrict__ denom,
                         float* __restrict__ aggout) {
    int e = blockIdx.x;                 // one edge per block
    int c = threadIdx.x;                // 0..255
    int src = ei[e], dst = ei[E_EDGES + e];
    float w = ew[e];
    int h = c >> 5;
    float wgt = alpha[(size_t)e * HEADS + h] /
                (denom[(size_t)dst * HEADS + h] + 1e-16f);
    float val = wgt * (v[(size_t)src * DIM + c] + w * We[c]);
    atomicAdd(&aggout[(size_t)dst * DIM + c], val);
}

// ----- beta gate + combine: one wave per node ------------------------------
__global__ void combine(const float* __restrict__ aggout, const float* __restrict__ xr,
                        const float* __restrict__ Wb, float* __restrict__ hout) {
    int lane = threadIdx.x & 31;
    int node = blockIdx.x * 8 + (threadIdx.x >> 5);
    const float* op = aggout + (size_t)node * DIM;
    const float* rp = xr     + (size_t)node * DIM;
    float s = 0.f;
#pragma unroll
    for (int i = 0; i < 8; ++i) {
        int c = lane + i * 32;
        float o = op[c], r = rp[c];
        s += o * Wb[c] + r * Wb[DIM + c] + (o - r) * Wb[2 * DIM + c];
    }
#pragma unroll
    for (int off = 16; off > 0; off >>= 1) s += __shfl_xor(s, off, 32);
    float beta = 1.f / (1.f + __expf(-s));
    float* hp = hout + (size_t)node * DIM;
#pragma unroll
    for (int i = 0; i < 8; ++i) {
        int c = lane + i * 32;
        float o = op[c], r = rp[c];
        hp[c] = beta * r + (1.f - beta) * o;
    }
}

// ---------------------------------------------------------------------------
extern "C" void kernel_launch(void* const* d_in, const int* in_sizes, int n_in,
                              void* d_out, int out_size, void* d_ws, size_t ws_size,
                              hipStream_t stream) {
    const float* x   = (const float*)d_in[0];
    const int*   ei  = (const int*)d_in[1];
    const float* ew  = (const float*)d_in[2];
    // param sets: p1 at 3..12, p2 at 13..22  (Wq,bq,Wk,bk,Wv,bv,We,Wskip,bskip,Wbeta)
    const float* P[2][10];
    for (int l = 0; l < 2; ++l)
        for (int j = 0; j < 10; ++j)
            P[l][j] = (const float*)d_in[3 + l * 10 + j];

    // ---- workspace carve (256B aligned) ----
    char* base = (char*)d_ws;
    size_t off = 0;
    auto carve = [&](size_t bytes) { void* p = base + off; off += (bytes + 255) & ~size_t(255); return p; };
    _Float16* h16    = (_Float16*)carve((size_t)N_NODES * DIM * 2);
    _Float16* wpack  = (_Float16*)carve((size_t)8 * 128 * 512 * 2);
    float*    bpack  = (float*)carve((size_t)8 * 256 * 4);
    float*    qkvxr  = (float*)carve((size_t)4 * N_NODES * DIM * 4);  // stacked q,k,v,xr
    float*    aggout = (float*)carve((size_t)N_NODES * DIM * 4);
    float*    alpha  = (float*)carve((size_t)E_EDGES * HEADS * 4);
    unsigned* mkey   = (unsigned*)carve((size_t)N_NODES * HEADS * 4);
    float*    denom  = (float*)carve((size_t)N_NODES * HEADS * 4);
    float*    hbuf   = (float*)carve((size_t)N_NODES * DIM * 4);

    float* qb  = qkvxr;
    float* kb  = qkvxr + (size_t)1 * N_NODES * DIM;
    float* vb  = qkvxr + (size_t)2 * N_NODES * DIM;
    float* xrb = qkvxr + (size_t)3 * N_NODES * DIM;

    // ---- pack all 8 weight matrices + biases once (layer1 set, shared set)
    Ptrs8 wps, bps;
    wps.p[0] = P[0][0]; wps.p[1] = P[0][2]; wps.p[2] = P[0][4]; wps.p[3] = P[0][7];
    wps.p[4] = P[1][0]; wps.p[5] = P[1][2]; wps.p[6] = P[1][4]; wps.p[7] = P[1][7];
    bps.p[0] = P[0][1]; bps.p[1] = P[0][3]; bps.p[2] = P[0][5]; bps.p[3] = P[0][8];
    bps.p[4] = P[1][1]; bps.p[5] = P[1][3]; bps.p[6] = P[1][5]; bps.p[7] = P[1][8];
    pack_w<<<(8 * 65536) / 256, 256, 0, stream>>>(wps, wpack);
    pack_bias<<<8, 256, 0, stream>>>(bps, bpack);

    const int nEH = E_EDGES * HEADS;
    for (int L = 0; L < 3; ++L) {
        int pi = (L == 0) ? 0 : 1;
        const float* hin  = (L == 0) ? x : hbuf;
        float*       hdst = (L == 2) ? (float*)d_out : hbuf;
        const _Float16* wpk = wpack + (size_t)pi * 4 * 128 * 512;
        const float*    bpk = bpack + (size_t)pi * 4 * 256;

        to_f16<<<(N_NODES * DIM) / 256, 256, 0, stream>>>(hin, h16);
        gemm4<<<ROWTILES, 256, LDS_BYTES, stream>>>(h16, wpk, bpk, qkvxr);
        init_layer<<<(N_NODES * DIM) / 256, 256, 0, stream>>>(aggout, mkey, denom);
        edge_alpha<<<nEH / 256, 256, 0, stream>>>(ei, ew, qb, kb, P[pi][6], alpha, mkey);
        edge_exp<<<nEH / 256, 256, 0, stream>>>(ei, alpha, mkey, denom);
        edge_agg<<<E_EDGES, 256, 0, stream>>>(ei, ew, vb, P[pi][6], alpha, denom, aggout);
        combine<<<N_NODES / 8, 256, 0, stream>>>(aggout, xrb, P[pi][9], hdst);
    }
}
